// Attention_41240275976828
// MI455X (gfx1250) — compile-verified, matrix operand order checked by reading, not run on hardware
//
#include <hip/hip_runtime.h>
#include <hip/hip_bf16.h>

typedef __bf16 bf16_t;
typedef __bf16 v16bf __attribute__((ext_vector_type(16)));
typedef __bf16 v8bf  __attribute__((ext_vector_type(8)));
typedef float  v8f   __attribute__((ext_vector_type(8)));

#define NH   16
#define DH   128
#define BATCH 2
#define SEQ  2048
#define HID  2048

union FragB { v16bf v; v8bf h[2]; };

__device__ __forceinline__ v8f vzero8() {
  v8f z;
#pragma unroll
  for (int i = 0; i < 8; ++i) z[i] = 0.0f;
  return z;
}

__device__ __forceinline__ v8f wmma_bf16(const FragB& a, const FragB& b, v8f c) {
  return __builtin_amdgcn_wmma_f32_16x16x32_bf16(false, a.v, false, b.v,
                                                 (short)0, c, false, false);
}

// Async global -> LDS copy of 16 bytes (per lane), tracked by ASYNCcnt.
__device__ __forceinline__ void async_copy_b128(void* lds_dst, const void* gsrc) {
  unsigned loff = (unsigned)(size_t)lds_dst;      // LDS aperture: addr[31:0] = LDS offset
  asm volatile("global_load_async_to_lds_b128 %0, %1, off"
               :: "v"(loff), "v"(gsrc) : "memory");
}
__device__ __forceinline__ void wait_asynccnt0() {
  asm volatile("s_wait_asynccnt 0" ::: "memory");
}

// ---------------------------------------------------------------------------
// f32 -> bf16 conversion, 8 elements / thread
// ---------------------------------------------------------------------------
__global__ void k_cvt8(const float* __restrict__ in, bf16_t* __restrict__ out, int n) {
  int i = (blockIdx.x * blockDim.x + threadIdx.x) * 8;
  if (i >= n) return;
  float4 f0 = *(const float4*)(in + i);
  float4 f1 = *(const float4*)(in + i + 4);
  v8bf o;
  o[0] = (bf16_t)f0.x; o[1] = (bf16_t)f0.y; o[2] = (bf16_t)f0.z; o[3] = (bf16_t)f0.w;
  o[4] = (bf16_t)f1.x; o[5] = (bf16_t)f1.y; o[6] = (bf16_t)f1.z; o[7] = (bf16_t)f1.w;
  *(v8bf*)(out + i) = o;
}

// ---------------------------------------------------------------------------
// Tiled bf16 GEMM: C[M,N] = A[M,K] * B[N,K]^T   (B given row-major N x K)
// Block: 256 threads (8 waves), tile 128x128, K-chunk 32.
// Waves arranged 4(M) x 2(N); each wave computes 32x64 = 8 WMMA accumulators.
// Staging via GLOBAL_LOAD_ASYNC_TO_LDS_B128 (ASYNCcnt), no VGPR landing zone.
// ---------------------------------------------------------------------------
#define GM 128
#define GN 128
#define GK 32

__global__ __launch_bounds__(256)
void k_gemm(const bf16_t* __restrict__ A, const bf16_t* __restrict__ Bm,
            void* __restrict__ outp, int M, int N, int K, int outF32)
{
  __shared__ bf16_t As[GM][GK];   // 8 KB
  __shared__ bf16_t Bs[GN][GK];   // 8 KB

  const int t    = threadIdx.x;
  const int lane = t & 31;
  const int w    = t >> 5;
  const int wm   = w >> 1;      // 0..3
  const int wn   = w & 1;       // 0..1
  const int bM   = blockIdx.y * GM;
  const int bN   = blockIdx.x * GN;
  const int lrow = lane & 15;
  const int lh   = lane >> 4;   // 0/1
  const int kb0  = lh * 8;

  v8f acc[2][4];
#pragma unroll
  for (int i = 0; i < 2; ++i)
#pragma unroll
    for (int j = 0; j < 4; ++j) acc[i][j] = vzero8();

  for (int k0 = 0; k0 < K; k0 += GK) {
    // stage A and B: each 128x32 bf16 = 8KB -> two async b128 copies per thread
#pragma unroll
    for (int l = 0; l < 2; ++l) {
      int idx = t + 256 * l;
      int r = idx >> 2, c = (idx & 3) << 3;
      async_copy_b128(&As[r][c], &A[(size_t)(bM + r) * K + k0 + c]);
      async_copy_b128(&Bs[r][c], &Bm[(size_t)(bN + r) * K + k0 + c]);
    }
    wait_asynccnt0();
    __syncthreads();

    FragB af[2], bfr[4];
#pragma unroll
    for (int mt = 0; mt < 2; ++mt) {
      const bf16_t* p = &As[wm * 32 + mt * 16 + lrow][kb0];
      af[mt].h[0] = *(const v8bf*)p;
      af[mt].h[1] = *(const v8bf*)(p + 16);
    }
#pragma unroll
    for (int nt = 0; nt < 4; ++nt) {
      const bf16_t* p = &Bs[wn * 64 + nt * 16 + lrow][kb0];
      bfr[nt].h[0] = *(const v8bf*)p;
      bfr[nt].h[1] = *(const v8bf*)(p + 16);
    }
#pragma unroll
    for (int mt = 0; mt < 2; ++mt)
#pragma unroll
      for (int nt = 0; nt < 4; ++nt)
        acc[mt][nt] = wmma_bf16(af[mt], bfr[nt], acc[mt][nt]);
    __syncthreads();
  }

  // C/D layout: lane holds col n = lane%16, rows v + 8*(lane/16)
#pragma unroll
  for (int mt = 0; mt < 2; ++mt)
#pragma unroll
    for (int nt = 0; nt < 4; ++nt) {
      int col = bN + wn * 64 + nt * 16 + lrow;
#pragma unroll
      for (int v = 0; v < 8; ++v) {
        int row = bM + wm * 32 + mt * 16 + v + 8 * lh;
        float val = acc[mt][nt][v];
        if (outF32) ((float*)outp)[(size_t)row * N + col] = val;
        else        ((bf16_t*)outp)[(size_t)row * N + col] = (bf16_t)val;
      }
    }
}

// ---------------------------------------------------------------------------
// RoPE + repack: qkv [B,S,3H] bf16 -> Q,K [B,NH,S,DH] (rotated), V^T [B,NH,DH,S]
// ---------------------------------------------------------------------------
__global__ __launch_bounds__(256)
void k_rope_pack(const bf16_t* __restrict__ qkv,
                 bf16_t* __restrict__ Qb, bf16_t* __restrict__ Kb,
                 bf16_t* __restrict__ Vt)
{
  int gid = blockIdx.x * 4 + (threadIdx.x >> 6);   // over B*NH*SEQ
  int d   = threadIdx.x & 63;
  int s   = gid % SEQ;
  int bh  = gid / SEQ;
  int h   = bh % NH;
  int b   = bh / NH;

  const bf16_t* base = qkv + ((size_t)(b * SEQ + s)) * (3 * HID);
  float q1 = (float)base[h * DH + d];
  float q2 = (float)base[h * DH + d + 64];
  float k1 = (float)base[HID + h * DH + d];
  float k2 = (float)base[HID + h * DH + d + 64];
  float v1 = (float)base[2 * HID + h * DH + d];
  float v2 = (float)base[2 * HID + h * DH + d + 64];

  float freq = __powf(10000.0f, -(float)(2 * d) / 128.0f);
  float ang  = (float)s * freq;
  float c = __cosf(ang), sn = __sinf(ang);

  size_t qkb = ((size_t)bh * SEQ + s) * DH;
  Qb[qkb + d]      = (bf16_t)(q1 * c - q2 * sn);
  Qb[qkb + d + 64] = (bf16_t)(q2 * c + q1 * sn);
  Kb[qkb + d]      = (bf16_t)(k1 * c - k2 * sn);
  Kb[qkb + d + 64] = (bf16_t)(k2 * c + k1 * sn);

  size_t vtb = (size_t)bh * DH * SEQ;
  Vt[vtb + (size_t)d * SEQ + s]        = (bf16_t)v1;
  Vt[vtb + (size_t)(d + 64) * SEQ + s] = (bf16_t)v2;
}

// ---------------------------------------------------------------------------
// Causal flash attention. One wave owns one 16-row q-tile; streams 32-key
// chunks. S^T = K*Q^T (softmax k-dim in per-lane VGPRs), O^T = V^T * P^T
// (per-lane online-softmax correction). P transits LDS for layout conversion.
// ---------------------------------------------------------------------------
__global__ __launch_bounds__(256)
void k_flash(const bf16_t* __restrict__ Qb, const bf16_t* __restrict__ Kb,
             const bf16_t* __restrict__ Vt, bf16_t* __restrict__ attnb)
{
  __shared__ bf16_t plds[8][16][32];   // per-wave P tile [q][k]
  const int lane = threadIdx.x & 31;
  const int w    = threadIdx.x >> 5;
  const int qt   = blockIdx.x * 8 + w;      // 0..127 q-tiles
  const int h    = blockIdx.y % NH;
  const int b    = blockIdx.y / NH;
  const int lq   = lane & 15;
  const int lh   = lane >> 4;
  const int kb0  = lh * 8;

  const size_t bh = (size_t)(b * NH + h);
  const bf16_t* Qbase = Qb + (bh * SEQ + (size_t)qt * 16) * DH;
  const bf16_t* Kbase = Kb + bh * SEQ * DH;
  const bf16_t* Vbase = Vt + bh * DH * SEQ;

  FragB qf[4];
#pragma unroll
  for (int s = 0; s < 4; ++s) {
    const bf16_t* p = Qbase + (size_t)lq * DH + 32 * s + kb0;
    qf[s].h[0] = *(const v8bf*)p;
    qf[s].h[1] = *(const v8bf*)(p + 16);
  }

  v8f o[8];
#pragma unroll
  for (int dt = 0; dt < 8; ++dt) o[dt] = vzero8();
  float m_run = -INFINITY, l_run = 0.0f;
  const float scale = 0.08838834764831845f;   // 1/sqrt(128)
  const int qglob = qt * 16 + lq;

  for (int kbase = 0; kbase < (qt + 1) * 16; kbase += 32) {
    v8f st0 = vzero8(), st1 = vzero8();
#pragma unroll
    for (int s = 0; s < 4; ++s) {
      FragB ka0, ka1;
      const bf16_t* p0 = Kbase + (size_t)(kbase + lq) * DH + 32 * s + kb0;
      ka0.h[0] = *(const v8bf*)p0;          ka0.h[1] = *(const v8bf*)(p0 + 16);
      const bf16_t* p1 = p0 + (size_t)16 * DH;
      ka1.h[0] = *(const v8bf*)p1;          ka1.h[1] = *(const v8bf*)(p1 + 16);
      st0 = wmma_bf16(ka0, qf[s], st0);
      st1 = wmma_bf16(ka1, qf[s], st1);
    }

    float cm = -INFINITY;
#pragma unroll
    for (int v = 0; v < 8; ++v) {
      int k0g = kbase + v + kb0;        // tile0 k row = v + 8*lh
      int k1g = k0g + 16;
      float s0 = (k0g <= qglob) ? st0[v] * scale : -INFINITY;
      float s1 = (k1g <= qglob) ? st1[v] * scale : -INFINITY;
      st0[v] = s0; st1[v] = s1;
      cm = fmaxf(cm, fmaxf(s0, s1));
    }
    cm = fmaxf(cm, __shfl_xor(cm, 16, 32));
    float m_new = fmaxf(m_run, cm);
    float corr  = __expf(m_run - m_new);

    float ls = 0.0f;
#pragma unroll
    for (int v = 0; v < 8; ++v) {
      float p0 = __expf(st0[v] - m_new);
      float p1 = __expf(st1[v] - m_new);
      ls += p0 + p1;
      plds[w][lq][v + kb0]      = (bf16_t)p0;
      plds[w][lq][16 + v + kb0] = (bf16_t)p1;
    }
    ls += __shfl_xor(ls, 16, 32);
    l_run = l_run * corr + ls;
    m_run = m_new;

#pragma unroll
    for (int dt = 0; dt < 8; ++dt)
#pragma unroll
      for (int e = 0; e < 8; ++e) o[dt][e] *= corr;

    asm volatile("s_wait_dscnt 0" ::: "memory");

    FragB pf;   // B-operand: lane holds P row q = lq
    pf.h[0] = *(const v8bf*)&plds[w][lq][kb0];
    pf.h[1] = *(const v8bf*)&plds[w][lq][kb0 + 16];
#pragma unroll
    for (int dt = 0; dt < 8; ++dt) {
      FragB vf; // A-operand: lane holds V^T row d = dt*16+lq, k = kb0..kb0+7, kb0+16..kb0+23
      const bf16_t* vp = Vbase + (size_t)(dt * 16 + lq) * SEQ + kbase + kb0;
      vf.h[0] = *(const v8bf*)vp;
      vf.h[1] = *(const v8bf*)(vp + 16);
      o[dt] = wmma_bf16(vf, pf, o[dt]);
    }
  }

  float inv = 1.0f / l_run;
  bf16_t* outr = attnb + ((size_t)b * SEQ + qglob) * HID + h * DH;
#pragma unroll
  for (int dt = 0; dt < 8; ++dt)
#pragma unroll
    for (int v = 0; v < 8; ++v)
      outr[dt * 16 + v + 8 * lh] = (bf16_t)(o[dt][v] * inv);
}

// ---------------------------------------------------------------------------
extern "C" void kernel_launch(void* const* d_in, const int* in_sizes, int n_in,
                              void* d_out, int out_size, void* d_ws, size_t ws_size,
                              hipStream_t stream) {
  const float* x     = (const float*)d_in[0];   // [2,2048,2048]
  const float* w_qkv = (const float*)d_in[1];   // [6144,2048]
  const float* w_o   = (const float*)d_in[2];   // [2048,2048]
  float* out = (float*)d_out;

  const int M = BATCH * SEQ;        // 4096
  const int N3 = 3 * HID;           // 6144

  char* ws = (char*)d_ws;
  size_t off = 0;
  bf16_t* xb    = (bf16_t*)(ws + off); off += (size_t)M * HID * 2;
  bf16_t* wqkvb = (bf16_t*)(ws + off); off += (size_t)N3 * HID * 2;
  bf16_t* wob   = (bf16_t*)(ws + off); off += (size_t)HID * HID * 2;
  bf16_t* qkvb  = (bf16_t*)(ws + off); off += (size_t)M * N3 * 2;
  bf16_t* Qb    = (bf16_t*)(ws + off); off += (size_t)M * HID * 2;
  bf16_t* Kb    = (bf16_t*)(ws + off); off += (size_t)M * HID * 2;
  bf16_t* Vt    = (bf16_t*)(ws + off); off += (size_t)M * HID * 2;
  bf16_t* attnb = (bf16_t*)(ws + off); off += (size_t)M * HID * 2;

  // 1) convert inputs to bf16
  {
    int n = M * HID;
    k_cvt8<<<n / 8 / 256, 256, 0, stream>>>(x, xb, n);
    n = N3 * HID;
    k_cvt8<<<n / 8 / 256, 256, 0, stream>>>(w_qkv, wqkvb, n);
    n = HID * HID;
    k_cvt8<<<n / 8 / 256, 256, 0, stream>>>(w_o, wob, n);
  }

  // 2) QKV projection: [4096,6144] = xb[4096,2048] * wqkvb[6144,2048]^T
  k_gemm<<<dim3(N3 / GN, M / GM), 256, 0, stream>>>(xb, wqkvb, qkvb, M, N3, HID, 0);

  // 3) RoPE + head repack (V transposed)
  k_rope_pack<<<(BATCH * NH * SEQ) / 4, 256, 0, stream>>>(qkvb, Qb, Kb, Vt);

  // 4) causal flash attention: 128 q-tiles / 8 waves per block, 32 (b,h) pairs
  k_flash<<<dim3(SEQ / 16 / 8, BATCH * NH), 256, 0, stream>>>(Qb, Kb, Vt, attnb);

  // 5) output projection (f32 out): [4096,2048] = attnb * wob[2048,2048]^T
  k_gemm<<<dim3(HID / GN, M / GM), 256, 0, stream>>>(attnb, wob, out, M, HID, HID, 1);
}
